// SS2D_Mamba_29085518528518
// MI455X (gfx1250) — compile-verified, matrix operand order checked by reading, not run on hardware
//
#include <hip/hip_runtime.h>
#include <hip/hip_bf16.h>
#include <hip/hip_fp16.h>

// ---------------------------------------------------------------------------
// SS2D bidirectional Mamba for MI455X (gfx1250, wave32, WMMA)
// Shapes: B=4, C=192, H=W=64, L=4096, BL=16384, DI=384, D_STATE=16, DTR=12
// ---------------------------------------------------------------------------

#define BB      4
#define CC      192
#define LL      4096
#define BL      16384
#define DI      384
#define DSTATE  16
#define DTR     12
#define NXP     44          // DTR + 2*DSTATE
#define NXP_PAD 48          // padded to 16
#define KDT_PAD 32          // dt-proj K padded 12 -> 32
#define NIN     768         // 2*DI
#define NC      64          // scan chunks per sequence
#define CH      64          // steps per chunk (NC*CH == LL)

typedef __attribute__((ext_vector_type(16))) _Float16 v16h;
typedef __attribute__((ext_vector_type(8)))  _Float16 v8h;
typedef __attribute__((ext_vector_type(8)))  float    v8f;

// ---------------------------------------------------------------------------
// Register-blocked WMMA GEMM: C[M,N] = A[M,K] * W[N,K]^T.
// K is a template parameter so all fragment offsets fold into the load's
// 24-bit immediate field: only two live base pointers -> no spills.
// Fragment layout (ISA 7.12.2): lane&15 = row, kbase = 8*(lane>>4);
// halves at K=kbase and K=kbase+16.  A and W fragments load identically.
// mode 0: plain f32 store (ldc stride)
// mode 1: f32 store of softplus(acc + bias[n])    (dt projection)
// mode 2: scatter into (B,C,H,W) output: row m = b*4096+hw, col n = channel
// ---------------------------------------------------------------------------
template <int MT, int NT, int K>
__global__ void wmma_gemm_tiled(const _Float16* __restrict__ A,
                                const _Float16* __restrict__ W,
                                float* __restrict__ C,
                                const float* __restrict__ bias,
                                int M, int N, int ldc, int mode) {
  const int lane = threadIdx.x & 31;
  const int wave = blockIdx.x * (blockDim.x >> 5) + (threadIdx.x >> 5);
  const int nMacro = N / (16 * NT);
  const int mMacro = M / (16 * MT);
  if (wave >= nMacro * mMacro) return;             // wave-uniform: EXEC stays full
  const int mt = wave / nMacro;
  const int nt = wave - mt * nMacro;

  const int row   = lane & 15;
  const int kbase = (lane >> 4) << 3;
  const _Float16* pA = A + (size_t)(mt * MT * 16) * K + row * K + kbase;
  const _Float16* pW = W + (size_t)(nt * NT * 16) * K + row * K + kbase;

  v8f acc[MT][NT] = {};
#pragma unroll 2
  for (int k = 0; k < K; k += 32) {
    v16h af[MT], bf[NT];
#pragma unroll
    for (int i = 0; i < MT; ++i) {
      v8h lo = *(const v8h*)(pA + i * 16 * K + k);
      v8h hi = *(const v8h*)(pA + i * 16 * K + k + 16);
      af[i] = __builtin_shufflevector(lo, hi, 0,1,2,3,4,5,6,7,8,9,10,11,12,13,14,15);
    }
#pragma unroll
    for (int j = 0; j < NT; ++j) {
      v8h lo = *(const v8h*)(pW + j * 16 * K + k);
      v8h hi = *(const v8h*)(pW + j * 16 * K + k + 16);
      bf[j] = __builtin_shufflevector(lo, hi, 0,1,2,3,4,5,6,7,8,9,10,11,12,13,14,15);
    }
#pragma unroll
    for (int i = 0; i < MT; ++i)
#pragma unroll
      for (int j = 0; j < NT; ++j)
        acc[i][j] = __builtin_amdgcn_wmma_f32_16x16x32_f16(
            false, af[i], false, bf[j], (short)0, acc[i][j], false, false);
  }

#pragma unroll
  for (int i = 0; i < MT; ++i) {
    const int m0 = ((mt * MT + i) << 4) + ((lane >> 4) << 3);  // VGPR r -> row m0+r
#pragma unroll
    for (int j = 0; j < NT; ++j) {
      const int n = ((nt * NT + j) << 4) + (lane & 15);
      if (mode == 0) {
#pragma unroll
        for (int r = 0; r < 8; ++r)
          C[(size_t)(m0 + r) * ldc + n] = acc[i][j][r];
      } else if (mode == 1) {
        const float bv = bias[n];
#pragma unroll
        for (int r = 0; r < 8; ++r) {
          float v = acc[i][j][r] + bv;
          float sp = (v > 20.f) ? v : __logf(1.f + __expf(v));   // softplus
          C[(size_t)(m0 + r) * ldc + n] = sp;
        }
      } else {                                      // scatter to (B,C,H,W)
#pragma unroll
        for (int r = 0; r < 8; ++r) {
          int m  = m0 + r;
          int b  = m >> 12;
          int hw = m & 4095;
          C[(size_t)((b * N + n) << 12) + hw] = acc[i][j][r];
        }
      }
    }
  }
}

// ---------------------------------------------------------------------------
// LayerNorm over channels: x (B,C,H,W) f32 -> x_seq (BL, C) f16.
// One wave per spatial position; each lane reduces 6 channels.
// ---------------------------------------------------------------------------
__global__ void ln_kernel(const float* __restrict__ x,
                          const float* __restrict__ w,
                          const float* __restrict__ bvec,
                          _Float16* __restrict__ xseq) {
  const int row  = blockIdx.x * (blockDim.x >> 5) + (threadIdx.x >> 5);
  const int lane = threadIdx.x & 31;
  const int b = row >> 12, hw = row & 4095;
  const float* xb = x + (size_t)b * CC * LL + hw;

  float v[6];
  float sum = 0.f;
#pragma unroll
  for (int i = 0; i < 6; ++i) {
    int c = lane + 32 * i;
    v[i] = xb[(size_t)c * LL];
    sum += v[i];
  }
#pragma unroll
  for (int off = 16; off; off >>= 1) sum += __shfl_xor(sum, off);
  const float mu = sum * (1.f / CC);

  float var = 0.f;
#pragma unroll
  for (int i = 0; i < 6; ++i) { float d = v[i] - mu; var += d * d; }
#pragma unroll
  for (int off = 16; off; off >>= 1) var += __shfl_xor(var, off);
  const float rstd = rsqrtf(var * (1.f / CC) + 1e-5f);

#pragma unroll
  for (int i = 0; i < 6; ++i) {
    int c = lane + 32 * i;
    float xn = (v[i] - mu) * rstd * w[c] + bvec[c];
    xseq[(size_t)row * CC + c] = (_Float16)xn;
  }
}

// ---------------------------------------------------------------------------
// Depthwise causal conv (k=4) along the sequence + SiLU, per direction.
// dir=0: taps l-3..l ; dir=1 (flipped sequence): taps l..l+3 reversed kernel.
// ---------------------------------------------------------------------------
__global__ void conv_silu_kernel(const float* __restrict__ xz,
                                 const float* __restrict__ cw,
                                 const float* __restrict__ cb,
                                 float* __restrict__ xs32,
                                 _Float16* __restrict__ xs16, int dir) {
  const int idx = blockIdx.x;        // 0..BL-1
  const int d   = threadIdx.x;       // 0..383
  const int b = idx >> 12, l = idx & 4095;
  float acc = cb[d];
#pragma unroll
  for (int j = 0; j < 4; ++j) {
    int lp = dir ? (l + 3 - j) : (l - 3 + j);
    if (lp >= 0 && lp < LL)
      acc += cw[d * 4 + j] * xz[(size_t)((b << 12) + lp) * NIN + d];
  }
  const float s = acc / (1.f + __expf(-acc));      // SiLU
  xs32[(size_t)idx * DI + d] = s;
  xs16[(size_t)idx * DI + d] = (_Float16)s;
}

// ---------------------------------------------------------------------------
// Blocked selective scan (diagonal decay => exact chunking).
// Scan-order step u in [0,4096): position l = dir ? 4095-u : u.
// Chunk c owns u in [c*CH, (c+1)*CH).
// pass1: local scan with h0 = 0 -> chunk state q[b,c,d,16], dt-sum S[b,c,d].
// combine: per (b,d) lane walks 64 chunks: hstart[c] = h; h = exp(a*S)h + q.
// pass2: re-run chunk seeded with hstart, emit G = (y + x*D) * silu(z).
// ---------------------------------------------------------------------------
__global__ void scan_pass1(const float* __restrict__ dt,
                           const float* __restrict__ xs,
                           const float* __restrict__ dbl,   // (BL,48): B@12
                           const float* __restrict__ Aneg,  // (DI,16)
                           float* __restrict__ S,
                           float* __restrict__ q, int dir) {
  const int d = blockIdx.x * blockDim.x + threadIdx.x;   // 0..383
  const int c = blockIdx.y;                              // chunk
  const int b = blockIdx.z;

  float h[DSTATE], a[DSTATE];
#pragma unroll
  for (int s = 0; s < DSTATE; ++s) { h[s] = 0.f; a[s] = Aneg[d * DSTATE + s]; }
  float Ssum = 0.f;

  for (int u = c * CH; u < c * CH + CH; ++u) {
    const int l   = dir ? (LL - 1 - u) : u;
    const int idx = (b << 12) + l;
    const float dtv = dt[(size_t)idx * DI + d];
    const float xv  = xs[(size_t)idx * DI + d];
    const float dx  = dtv * xv;
    const float* Bp = dbl + (size_t)idx * NXP_PAD + DTR;
    Ssum += dtv;
#pragma unroll
    for (int s = 0; s < DSTATE; ++s)
      h[s] = h[s] * __expf(dtv * a[s]) + dx * Bp[s];
  }
  const size_t base = (size_t)(b * NC + c) * DI + d;
  S[base] = Ssum;
#pragma unroll
  for (int s = 0; s < DSTATE; ++s) q[base * DSTATE + s] = h[s];
}

__global__ void scan_combine(const float* __restrict__ S,
                             const float* __restrict__ q,
                             const float* __restrict__ Aneg,
                             float* __restrict__ hstart) {
  const int d = blockIdx.x * blockDim.x + threadIdx.x;   // 0..383
  const int b = blockIdx.y;
  float h[DSTATE], a[DSTATE];
#pragma unroll
  for (int s = 0; s < DSTATE; ++s) { h[s] = 0.f; a[s] = Aneg[d * DSTATE + s]; }
  for (int c = 0; c < NC; ++c) {
    const size_t base = (size_t)(b * NC + c) * DI + d;
    const float decay_arg = S[base];
#pragma unroll
    for (int s = 0; s < DSTATE; ++s) {
      hstart[base * DSTATE + s] = h[s];
      h[s] = h[s] * __expf(a[s] * decay_arg) + q[base * DSTATE + s];
    }
  }
}

__global__ void scan_pass2(const float* __restrict__ dt,
                           const float* __restrict__ xs,
                           const float* __restrict__ dbl,   // B@12, C@28
                           const float* __restrict__ xz,    // z at col 384
                           const float* __restrict__ Aneg,
                           const float* __restrict__ Dvec,
                           const float* __restrict__ hstart,
                           float* __restrict__ G, int dir, int accum) {
  const int d = blockIdx.x * blockDim.x + threadIdx.x;
  const int c = blockIdx.y;
  const int b = blockIdx.z;

  const size_t base = (size_t)(b * NC + c) * DI + d;
  float h[DSTATE], a[DSTATE];
#pragma unroll
  for (int s = 0; s < DSTATE; ++s) {
    h[s] = hstart[base * DSTATE + s];
    a[s] = Aneg[d * DSTATE + s];
  }
  const float Dv = Dvec[d];

  for (int u = c * CH; u < c * CH + CH; ++u) {
    const int l   = dir ? (LL - 1 - u) : u;
    const int idx = (b << 12) + l;
    const float dtv = dt[(size_t)idx * DI + d];
    const float xv  = xs[(size_t)idx * DI + d];
    const float dx  = dtv * xv;
    const float* Bp = dbl + (size_t)idx * NXP_PAD + DTR;
    const float* Cp = Bp + DSTATE;
    float y = 0.f;
#pragma unroll
    for (int s = 0; s < DSTATE; ++s) {
      float hs = h[s] * __expf(dtv * a[s]) + dx * Bp[s];
      h[s] = hs;
      y += hs * Cp[s];
    }
    const float zv = xz[(size_t)idx * NIN + DI + d];
    const float sz = zv / (1.f + __expf(-zv));
    const float g  = (y + xv * Dv) * sz;
    float* gp = G + (size_t)idx * DI + d;
    if (accum) *gp += g; else *gp = g;
  }
}

// ---------------------------------------------------------------------------
// Small prep / pack kernels
// ---------------------------------------------------------------------------
__global__ void cvt_f32_to_f16(const float* __restrict__ s,
                               _Float16* __restrict__ d, int n) {
  int i = blockIdx.x * blockDim.x + threadIdx.x;
  if (i < n) d[i] = (_Float16)s[i];
}

// x_proj weights (44,384) -> (48,384) f16, rows 44..47 zero
__global__ void pad_xproj_kernel(const float* __restrict__ s,
                                 _Float16* __restrict__ d) {
  int row = blockIdx.x, col = threadIdx.x;     // grid 48, block 384
  d[(size_t)row * DI + col] =
      (row < NXP) ? (_Float16)s[(size_t)row * DI + col] : (_Float16)0.f;
}

// dt_proj weights (384,12) -> (384,32) f16, cols 12..31 zero
__global__ void pad_dtproj_kernel(const float* __restrict__ s,
                                  _Float16* __restrict__ d) {
  int i = blockIdx.x * blockDim.x + threadIdx.x;
  if (i >= DI * KDT_PAD) return;
  int row = i >> 5, col = i & 31;
  d[i] = (col < DTR) ? (_Float16)s[row * DTR + col] : (_Float16)0.f;
}

// A = -exp(A_log)
__global__ void aneg_kernel(const float* __restrict__ alog,
                            float* __restrict__ an) {
  int i = blockIdx.x * blockDim.x + threadIdx.x;
  if (i < DI * DSTATE) an[i] = -__expf(alog[i]);
}

// dbl f32 (BL,48) -> padded f16 (BL,32): cols 0..11 copied, 12..31 zero
__global__ void pack_dbl_kernel(const float* __restrict__ s,
                                _Float16* __restrict__ d) {
  int i = blockIdx.x * blockDim.x + threadIdx.x;
  if (i >= BL * KDT_PAD) return;
  int row = i >> 5, col = i & 31;
  d[i] = (col < DTR) ? (_Float16)s[(size_t)row * NXP_PAD + col] : (_Float16)0.f;
}

// ---------------------------------------------------------------------------
// Host launcher
// ---------------------------------------------------------------------------
static inline size_t alignUp(size_t v) { return (v + 255) & ~(size_t)255; }

extern "C" void kernel_launch(void* const* d_in, const int* in_sizes, int n_in,
                              void* d_out, int out_size, void* d_ws, size_t ws_size,
                              hipStream_t stream) {
  (void)in_sizes; (void)n_in; (void)out_size; (void)ws_size;

  const float* x        = (const float*)d_in[0];
  const float* ln_w     = (const float*)d_in[1];
  const float* ln_b     = (const float*)d_in[2];
  const float* in_proj  = (const float*)d_in[3];
  const float* conv_w   = (const float*)d_in[4];
  const float* conv_b   = (const float*)d_in[5];
  const float* x_proj   = (const float*)d_in[6];
  const float* dt_proj  = (const float*)d_in[7];
  const float* dt_bias  = (const float*)d_in[8];
  const float* A_log    = (const float*)d_in[9];
  const float* D_vec    = (const float*)d_in[10];
  const float* out_proj = (const float*)d_in[11];
  float* out = (float*)d_out;

  char* ws = (char*)d_ws;
  size_t off = 0;
  auto alloc = [&](size_t bytes) { size_t o = off; off = alignUp(off + bytes); return ws + o; };

  _Float16* xseq16  = (_Float16*)alloc((size_t)BL * CC * 2);
  _Float16* w_in16  = (_Float16*)alloc((size_t)NIN * CC * 2);
  _Float16* w_xp16  = (_Float16*)alloc((size_t)NXP_PAD * DI * 2);
  _Float16* w_dt16  = (_Float16*)alloc((size_t)DI * KDT_PAD * 2);
  _Float16* w_out16 = (_Float16*)alloc((size_t)CC * DI * 2);
  float*    Aneg    = (float*)   alloc((size_t)DI * DSTATE * 4);
  float*    xz      = (float*)   alloc((size_t)BL * NIN * 4);
  float*    xs32    = (float*)   alloc((size_t)BL * DI * 4);
  _Float16* xs16    = (_Float16*)alloc((size_t)BL * DI * 2);
  float*    dbl     = (float*)   alloc((size_t)BL * NXP_PAD * 4);
  _Float16* dblpad  = (_Float16*)alloc((size_t)BL * KDT_PAD * 2);
  float*    dt32    = (float*)   alloc((size_t)BL * DI * 4);
  float*    G32     = (float*)   alloc((size_t)BL * DI * 4);
  _Float16* G16     = (_Float16*)alloc((size_t)BL * DI * 2);
  float*    Ssum    = (float*)   alloc((size_t)BB * NC * DI * 4);
  float*    qbuf    = (float*)   alloc((size_t)BB * NC * DI * DSTATE * 4);
  float*    hstart  = (float*)   alloc((size_t)BB * NC * DI * DSTATE * 4);

  // ---- weight prep ----
  cvt_f32_to_f16<<<(NIN * CC + 255) / 256, 256, 0, stream>>>(in_proj, w_in16, NIN * CC);
  cvt_f32_to_f16<<<(CC * DI + 255) / 256, 256, 0, stream>>>(out_proj, w_out16, CC * DI);
  pad_xproj_kernel<<<NXP_PAD, DI, 0, stream>>>(x_proj, w_xp16);
  pad_dtproj_kernel<<<(DI * KDT_PAD + 255) / 256, 256, 0, stream>>>(dt_proj, w_dt16);
  aneg_kernel<<<(DI * DSTATE + 255) / 256, 256, 0, stream>>>(A_log, Aneg);

  // ---- layernorm -> x_seq f16 ----
  ln_kernel<<<BL / 8, 256, 0, stream>>>(x, ln_w, ln_b, xseq16);

  // ---- in_proj GEMM: (BL,192) x (768,192)^T -> xz (BL,768) ----
  {
    int waves = (BL / 32) * (NIN / 64);            // (M/16/2) * (N/16/4)
    wmma_gemm_tiled<2, 4, CC><<<(waves + 7) / 8, 256, 0, stream>>>(
        xseq16, w_in16, xz, nullptr, BL, NIN, NIN, 0);
  }

  // ---- per-direction pipeline (buffers reused) ----
  for (int dir = 0; dir < 2; ++dir) {
    conv_silu_kernel<<<BL, DI, 0, stream>>>(xz, conv_w, conv_b, xs32, xs16, dir);

    // x_proj GEMM: (BL,384) x (48,384)^T -> dbl (BL,48)
    {
      int waves = (BL / 32) * (NXP_PAD / 48);      // (M/16/2) * (N/16/3)
      wmma_gemm_tiled<2, 3, DI><<<(waves + 7) / 8, 256, 0, stream>>>(
          xs16, w_xp16, dbl, nullptr, BL, NXP_PAD, NXP_PAD, 0);
    }

    pack_dbl_kernel<<<(BL * KDT_PAD + 255) / 256, 256, 0, stream>>>(dbl, dblpad);

    // dt_proj GEMM + bias + softplus: (BL,32) x (384,32)^T -> dt (BL,384)
    {
      int waves = (BL / 32) * (DI / 64);
      wmma_gemm_tiled<2, 4, KDT_PAD><<<(waves + 7) / 8, 256, 0, stream>>>(
          dblpad, w_dt16, dt32, dt_bias, BL, DI, DI, 1);
    }

    // ---- blocked selective scan ----
    scan_pass1<<<dim3(DI / 128, NC, BB), 128, 0, stream>>>(
        dt32, xs32, dbl, Aneg, Ssum, qbuf, dir);
    scan_combine<<<dim3(DI / 128, BB), 128, 0, stream>>>(Ssum, qbuf, Aneg, hstart);
    scan_pass2<<<dim3(DI / 128, NC, BB), 128, 0, stream>>>(
        dt32, xs32, dbl, xz, Aneg, D_vec, hstart, G32, dir, dir);
  }

  // ---- pack G and out_proj GEMM with transpose-scatter to (B,C,H,W) ----
  cvt_f32_to_f16<<<(BL * DI + 255) / 256, 256, 0, stream>>>(G32, G16, BL * DI);
  {
    int waves = (BL / 32) * (CC / 64);
    wmma_gemm_tiled<2, 4, DI><<<(waves + 7) / 8, 256, 0, stream>>>(
        G16, w_out16, out, nullptr, BL, CC, CC, 2);
  }
}
